// STCMaskSAM_88467736363640
// MI455X (gfx1250) — matrix operand order repaired and checked by reading, not compiled, not run on hardware
//
#include <hip/hip_runtime.h>
#include <stddef.h>
#include <stdint.h>

// ---------------------------------------------------------------------------
// Problem constants (from reference): B=4, N=M=48, H=192, W=640
// ---------------------------------------------------------------------------
#define BB      4
#define NN      48
#define MM      48
#define HH      192
#define WW      640
#define HWPIX   (HH * WW)          // 122880 pixels per mask
#define IOU_TH  0.8f

typedef unsigned long long ull;
typedef unsigned int       uint;

typedef __attribute__((ext_vector_type(16))) _Float16 v16h;
typedef __attribute__((ext_vector_type(8)))  _Float16 v8h;
typedef __attribute__((ext_vector_type(8)))  float    v8f;

// ---------------------------------------------------------------------------
// Workspace layout (bytes). Total ~7.9 MB.
//   abits : BB*HWPIX u64   (bit n = a_n(p) > 0)
//   bbits : BB*HWPIX u64
//   qrows : BB*NN    u64   (bit m = q[n][m])
//   inter : BB*NN*MM f32
//   sa,sb : BB*48    u32
// ---------------------------------------------------------------------------
#define OFF_ABITS 0
#define OFF_BBITS (OFF_ABITS + (size_t)BB * HWPIX * 8)            // 3,932,160
#define OFF_QROWS (OFF_BBITS + (size_t)BB * HWPIX * 8)            // 7,864,320
#define OFF_INTER (OFF_QROWS + (size_t)BB * NN * 8)               // 7,865,856
#define OFF_SA    (OFF_INTER + (size_t)BB * NN * MM * 4)          // 7,902,720
#define OFF_SB    (OFF_SA    + (size_t)BB * NN * 4)               // 7,903,488
#define ZERO_WORDS ((uint)((OFF_SB + (size_t)BB * MM * 4 - OFF_QROWS) / 4))  // 9984

// ---------------------------------------------------------------------------
// Kernel 0: zero the accumulator region (qrows | inter | sa | sb)
// ---------------------------------------------------------------------------
__global__ void stc_zero_kernel(uint* p, uint nwords) {
    for (uint i = threadIdx.x; i < nwords; i += blockDim.x) p[i] = 0u;
}

// ---------------------------------------------------------------------------
// Kernel 1: binarize + bit-pack both tensors; per-mask pixel counts via
// wave32 ballot (gfx1250 is wave32: EXEC/ballot are 32-bit wide).
// grid = (HWPIX/256, BB), block = 256
// ---------------------------------------------------------------------------
__global__ __launch_bounds__(256)
void stc_pack_kernel(const float* __restrict__ a, const float* __restrict__ b,
                     ull* __restrict__ abits, ull* __restrict__ bbits,
                     uint* __restrict__ sa, uint* __restrict__ sb) {
    const uint tid   = threadIdx.x;
    const uint lane  = tid & 31u;              // wave32
    const uint batch = blockIdx.y;
    const uint p     = blockIdx.x * 256u + tid;

    __shared__ uint s_sa[NN];
    __shared__ uint s_sb[MM];
    if (tid < NN) { s_sa[tid] = 0u; s_sb[tid] = 0u; }
    __syncthreads();

    const size_t base = (size_t)batch * NN * HWPIX + p;
    ull ab = 0ull, bbv = 0ull;
#pragma unroll 4
    for (int n = 0; n < NN; ++n) {
        const bool va = a[base + (size_t)n * HWPIX] > 0.0f;
        const bool vb = b[base + (size_t)n * HWPIX] > 0.0f;
        ab  |= (ull)va << n;
        bbv |= (ull)vb << n;
        const uint ca = (uint)__popcll(__ballot(va));   // pixels set in this wave
        const uint cb = (uint)__popcll(__ballot(vb));
        if (lane == 0u) {
            atomicAdd(&s_sa[n], ca);
            atomicAdd(&s_sb[n], cb);
        }
    }
    abits[(size_t)batch * HWPIX + p] = ab;
    bbits[(size_t)batch * HWPIX + p] = bbv;

    __syncthreads();
    if (tid < NN) {
        atomicAdd(&sa[batch * NN + tid], s_sa[tid]);
        atomicAdd(&sb[batch * MM + tid], s_sb[tid]);
    }
}

// ---------------------------------------------------------------------------
// Kernel 2: inter[b,n,m] = sum_p a_n(p)*b_m(p) via v_wmma_f32_16x16x32_f16.
// Products are 0/1 in f16 (exact); accumulation in f32 (counts < 2^24, exact).
//
// grid = (KBLOCKS, BB), block = 288 threads = 9 waves.
// Wave w owns output tile (ti = w/3, tj = w%3) of the 48x48 result and
// accumulates over this block's K-slice (KCHUNK pixels), atomicAdd at end.
// Floats are binarized to f16 into padded LDS; fragments follow the CDNA5
// 16-bit A (16x32) / B (32x16) VGPR layouts (ISA 7.12.2):
//   A: lane l -> row = l&15; halves 0..7  = K base..base+7,
//                             halves 8..15 = K base+16..base+23, base=(l<16?0:8)
//   B: lane l -> col = l&15; halves 0..7  = K base..base+7,
//                             halves 8..15 = K base+16..base+23, base=(l<16?0:8)
// ---------------------------------------------------------------------------
#define KBLOCKS 96
#define KCHUNK  (HWPIX / KBLOCKS)   // 1280
#define KSTEP   128                 // pixels staged in LDS per step
#define LDA     (KSTEP + 8)         // half-element row pitch: 272B -> spreads banks

__global__ __launch_bounds__(288)
void stc_inter_kernel(const float* __restrict__ a, const float* __restrict__ b,
                      float* __restrict__ inter) {
    __shared__ _Float16 As[NN * LDA];
    __shared__ _Float16 Bs[MM * LDA];

    const uint tid   = threadIdx.x;
    const uint lane  = tid & 31u;
    const uint wave  = tid >> 5;       // 0..8
    const uint ti    = wave / 3u;      // n-tile
    const uint tj    = wave % 3u;      // m-tile
    const uint batch = blockIdx.y;
    const uint k0    = blockIdx.x * KCHUNK;

    const size_t gbase = (size_t)batch * NN * HWPIX + k0;

    union Frag { v16h v; v8h h8[2]; };
    v8f c = {};

    for (uint ks = 0; ks < KCHUNK; ks += KSTEP) {
        // ---- cooperative staging: float4 loads, binarize -> f16, pack 4 halves
        for (uint i = tid; i < (NN * KSTEP) / 4; i += 288u) {
            const uint row = i / (KSTEP / 4);
            const uint col = (i % (KSTEP / 4)) * 4u;
            const size_t g = gbase + (size_t)row * HWPIX + ks + col;
            const float4 va = *(const float4*)(a + g);
            const float4 vb = *(const float4*)(b + g);
            union { _Float16 h[4]; ull u; } pa, pb;
            pa.h[0] = va.x > 0.f ? (_Float16)1 : (_Float16)0;
            pa.h[1] = va.y > 0.f ? (_Float16)1 : (_Float16)0;
            pa.h[2] = va.z > 0.f ? (_Float16)1 : (_Float16)0;
            pa.h[3] = va.w > 0.f ? (_Float16)1 : (_Float16)0;
            pb.h[0] = vb.x > 0.f ? (_Float16)1 : (_Float16)0;
            pb.h[1] = vb.y > 0.f ? (_Float16)1 : (_Float16)0;
            pb.h[2] = vb.z > 0.f ? (_Float16)1 : (_Float16)0;
            pb.h[3] = vb.w > 0.f ? (_Float16)1 : (_Float16)0;
            *(ull*)(&As[row * LDA + col]) = pa.u;
            *(ull*)(&Bs[row * LDA + col]) = pb.u;
        }
        __syncthreads();

        // ---- 4 WMMA sub-steps of K=32 each
        const uint arow = ti * 16u + (lane & 15u);
        const uint bcol = tj * 16u + (lane & 15u);
        const uint abase = (lane < 16u) ? 0u : 8u;
        const uint bbase = (lane < 16u) ? 0u : 8u;
#pragma unroll
        for (uint kk = 0; kk < 4u; ++kk) {
            Frag fa, fb;
            const _Float16* ap = &As[arow * LDA + kk * 32u + abase];
            const _Float16* bp = &Bs[bcol * LDA + kk * 32u + bbase];
            fa.h8[0] = *(const v8h*)(ap);          // K base..base+7
            fa.h8[1] = *(const v8h*)(ap + 16);     // K base+16..base+23
            fb.h8[0] = *(const v8h*)(bp);          // K base..base+7
            fb.h8[1] = *(const v8h*)(bp + 16);     // K base+16..base+23
            c = __builtin_amdgcn_wmma_f32_16x16x32_f16(
                    false, fa.v, false, fb.v, (short)0, c, false, false);
        }
        __syncthreads();
    }

    // ---- flush: C layout (ISA 7.12.2): VGPR r, lane l -> M = r + (l<16?0:8),
    //      N = l&15
    const uint mcol = tj * 16u + (lane & 15u);
    const uint radd = (lane < 16u) ? 0u : 8u;
#pragma unroll
    for (uint r = 0; r < 8u; ++r) {
        const uint nrow = ti * 16u + r + radd;
        atomicAdd(&inter[((size_t)batch * NN + nrow) * MM + mcol], c[r]);
    }
}

// ---------------------------------------------------------------------------
// Kernel 3: IoU threshold -> per-n bitmask of qualifying m's.
// Mirrors reference: q = inter / max(union,1) > 0.8 in f32 (all counts are
// exact integers in f32, so this matches JAX bit-for-bit on the compare input).
// ---------------------------------------------------------------------------
__global__ void stc_q_kernel(const float* __restrict__ inter,
                             const uint* __restrict__ sa,
                             const uint* __restrict__ sb,
                             ull* __restrict__ qrows) {
    for (uint i = threadIdx.x; i < BB * NN * MM; i += blockDim.x) {
        const uint b = i / (NN * MM);
        const uint r = i % (NN * MM);
        const uint n = r / MM;
        const uint m = r % MM;
        const float it = inter[i];
        const float un = (float)sa[b * NN + n] + (float)sb[b * MM + m] - it;
        if (it / fmaxf(un, 1.0f) > IOU_TH)
            atomicOr(&qrows[b * NN + n], 1ull << m);
    }
}

// ---------------------------------------------------------------------------
// Kernel 4: per pixel: occluded iff exists n with
//   qrow[n] & (a_n ? ~bbits : bbits) != 0   (== "qualifying pair with XOR set")
// grid = (HWPIX/256, BB), block = 256
// ---------------------------------------------------------------------------
__global__ __launch_bounds__(256)
void stc_out_kernel(const ull* __restrict__ abits, const ull* __restrict__ bbits,
                    const ull* __restrict__ qrows, float* __restrict__ out) {
    __shared__ ull qr[NN];
    const uint tid   = threadIdx.x;
    const uint batch = blockIdx.y;
    if (tid < NN) qr[tid] = qrows[batch * NN + tid];
    __syncthreads();

    const uint p  = blockIdx.x * 256u + tid;
    const size_t idx = (size_t)batch * HWPIX + p;
    const ull ab = abits[idx];
    const ull bv = bbits[idx];
    const ull nb = ~bv;

    bool occ = false;
#pragma unroll 8
    for (int n = 0; n < NN; ++n) {
        const ull q = qr[n];
        if (q) {
            const ull hit = ((ab >> n) & 1ull) ? (q & nb) : (q & bv);
            occ |= (hit != 0ull);
        }
    }
    out[idx] = occ ? 0.0f : 1.0f;
}

// ---------------------------------------------------------------------------
// Launch
// ---------------------------------------------------------------------------
extern "C" void kernel_launch(void* const* d_in, const int* in_sizes, int n_in,
                              void* d_out, int out_size, void* d_ws, size_t ws_size,
                              hipStream_t stream) {
    const float* a = (const float*)d_in[0];   // stereo_warped_target  [B,N,H,W] f32
    const float* b = (const float*)d_in[1];   // temporal_warped_target[B,M,H,W] f32
    float* out = (float*)d_out;               // [B,1,H,W] f32

    char*  ws    = (char*)d_ws;               // needs ~7.91 MB
    ull*   abits = (ull*)  (ws + OFF_ABITS);
    ull*   bbits = (ull*)  (ws + OFF_BBITS);
    ull*   qrows = (ull*)  (ws + OFF_QROWS);
    float* inter = (float*)(ws + OFF_INTER);
    uint*  sa    = (uint*) (ws + OFF_SA);
    uint*  sb    = (uint*) (ws + OFF_SB);

    stc_zero_kernel<<<1, 256, 0, stream>>>((uint*)(ws + OFF_QROWS), ZERO_WORDS);

    stc_pack_kernel<<<dim3(HWPIX / 256, BB), 256, 0, stream>>>(a, b, abits, bbits, sa, sb);

    stc_inter_kernel<<<dim3(KBLOCKS, BB), 288, 0, stream>>>(a, b, inter);

    stc_q_kernel<<<1, 256, 0, stream>>>(inter, sa, sb, qrows);

    stc_out_kernel<<<dim3(HWPIX / 256, BB), 256, 0, stream>>>(abits, bbits, qrows, out);
}
